// BasicBlock_52201032516123
// MI455X (gfx1250) — compile-verified, matrix operand order checked by reading, not run on hardware
//
#include <hip/hip_runtime.h>

typedef __attribute__((ext_vector_type(16))) __bf16 v16bf;
typedef __attribute__((ext_vector_type(8)))  __bf16 v8bf;
typedef __attribute__((ext_vector_type(8)))  float  v8f;

#define C_CH 256
#define HH 56
#define WW 56
#define HWP (HH*WW)        // 3136
#define NB 32
#define MTOT (NB*HWP)      // 100352 = 784 * 128
#define KK (C_CH*9)        // 2304

#define KC 64              // K elements staged per buffer
#define NSTAGE (KK/KC)     // 36
#define TWE 72             // padded LDS row stride (elements; 144B keeps b128
                           // 16B-aligned; 36-word bank stride -> conflict-free)
#define TILE_E (128*TWE)   // elements per LDS tile (A or B) = 9216
#define BUF_E  (2*TILE_E)  // elements per buffer (A tile + B tile)
#define SMEM_BYTES (3*BUF_E*2)   // triple buffered, bf16 = 110592 B

static __device__ __forceinline__ v16bf cat8(v8bf lo, v8bf hi) {
  return __builtin_shufflevector(lo, hi, 0,1,2,3,4,5,6,7,8,9,10,11,12,13,14,15);
}

// gfx1250 async copy: global -> LDS, 16B, tracked by ASYNCcnt.
static __device__ __forceinline__ void async_ld16(unsigned ldsByte,
                                                  const void* g) {
  asm volatile("global_load_async_to_lds_b128 %0, %1, off"
               :: "v"(ldsByte), "v"(g) : "memory");
}
// Steady state: two 8-instruction stages in flight; <=8 means oldest is done.
static __device__ __forceinline__ void wait_async_le8() {
  asm volatile("s_wait_asynccnt 8" ::: "memory");
}
static __device__ __forceinline__ void wait_async0() {
  asm volatile("s_wait_asynccnt 0" ::: "memory");
}

// Fold BN scale into conv weights; layout W'[o][tap][c] (bf16) so a B-fragment
// lane reads contiguous 16B K-runs. Also emit fused bias = beta - mean*scale.
__global__ __launch_bounds__(256) void prep_w(
    const float* __restrict__ w, const float* __restrict__ g,
    const float* __restrict__ b, const float* __restrict__ rm,
    const float* __restrict__ rv,
    __bf16* __restrict__ wb, float* __restrict__ bias) {
  int idx = blockIdx.x * 256 + threadIdx.x;
  if (idx < C_CH * KK) {
    int o   = idx / KK;
    int r   = idx % KK;
    int tap = r / C_CH;
    int c   = r % C_CH;
    int kh  = tap / 3, kw = tap % 3;
    float s = g[o] * rsqrtf(rv[o] + 1e-5f);
    wb[idx] = (__bf16)(w[((o * C_CH + c) * 3 + kh) * 3 + kw] * s);
  }
  if (idx < C_CH) {
    float s = g[idx] * rsqrtf(rv[idx] + 1e-5f);
    bias[idx] = b[idx] - rm[idx] * s;
  }
}

// fp32 NCHW -> bf16 NHWC; also (block 0) zero the halo scratch region.
__global__ __launch_bounds__(256) void cvt_x(const float* __restrict__ x,
                                             __bf16* __restrict__ xb,
                                             float* __restrict__ zpad) {
  if (blockIdx.x == 0 && threadIdx.x < 64) zpad[threadIdx.x] = 0.0f;
  size_t d  = (size_t)blockIdx.x * 256 + threadIdx.x;   // [0, MTOT*C_CH)
  int    c  = (int)(d & (C_CH - 1));
  size_t m  = d >> 8;                                    // pixel n*HW + hw
  int  nimg = (int)(m / HWP);
  int  hw   = (int)(m % HWP);
  xb[d] = (__bf16)x[((size_t)nimg * C_CH + c) * HWP + hw];
}

// Implicit-GEMM 3x3 conv (stride 1, pad 1) via v_wmma_f32_16x16x32_bf16.
// Triple-buffered async-to-LDS pipeline: 64-wide K stages, two stages in
// flight, 16 WMMAs per barrier.
// MODE 1: out = relu(acc + bias)          -> bf16 NHWC (intermediate)
// MODE 2: out = relu(acc + bias + resid)  -> fp32 NCHW (final)
template <int MODE>
__global__ __launch_bounds__(256) void conv_gemm(
    const __bf16* __restrict__ A,      // bf16 NHWC activations [MTOT][C_CH]
    const __bf16* __restrict__ Wb,     // bf16 weights [C_CH][KK]
    const float*  __restrict__ bias,   // fused bias [C_CH]
    const float*  __restrict__ resid,  // fp32 NCHW residual (MODE 2)
    const void*   __restrict__ zpad,   // 256B of zeros (halo rows)
    __bf16* __restrict__ outbf,        // MODE 1 destination
    float*  __restrict__ outf) {       // MODE 2 destination
  extern __shared__ __bf16 sm[];       // [3 bufs][A tile | B tile]

  const int lane  = threadIdx.x & 31;
  const int wave  = threadIdx.x >> 5;
  const int waveM = wave & 3;          // 4 waves along M
  const int waveN = wave >> 2;         // 2 waves along N
  const int rowBase = blockIdx.x * 128 + waveM * 32;
  const int colBase = blockIdx.y * 128 + waveN * 64;
  const int l16  = lane & 15;
  const int koff = (lane >> 4) << 3;   // 0 or 8: per-lane K sub-offset

  // ---- staging identities (per thread, fixed) ----
  const int tS   = threadIdx.x;
  const int rowS = tS >> 1;            // 0..127: tile row/col staged
  const int chkE = (tS & 1) * 32;      // 64B chunk within the 128B row
  const int rA   = blockIdx.x * 128 + rowS;
  const int wA   = rA % WW;
  const int hA   = (rA / WW) % HH;
  const __bf16* aSrc = A  + (size_t)rA * C_CH + chkE;
  const __bf16* bSrc = Wb + (size_t)(blockIdx.y * 128 + rowS) * KK + chkE;
  const unsigned aDstB = (unsigned)(rowS * TWE + chkE) * 2;            // bytes
  const unsigned bDstB = (unsigned)(TILE_E + rowS * TWE + chkE) * 2;   // bytes

  // ---- fragment LDS offsets (per lane, fixed; elements) ----
  unsigned aOffE[2], bOffE[4];
#pragma unroll
  for (int i = 0; i < 2; ++i)
    aOffE[i] = (unsigned)((waveM * 32 + i * 16 + l16) * TWE + koff);
#pragma unroll
  for (int j = 0; j < 4; ++j)
    bOffE[j] = (unsigned)(TILE_E + (waveN * 64 + j * 16 + l16) * TWE + koff);

  v8f acc[2][4];
#pragma unroll
  for (int i = 0; i < 2; ++i)
#pragma unroll
    for (int j = 0; j < 4; ++j) acc[i][j] = {};

  // ---- stage one 64-wide K chunk (tap = st/4, chunk = st%4) into a buffer ---
  auto stage_step = [&](int st, unsigned bufB) {
    const int tap = st >> 2, ks2 = st & 3;
    const int dh = tap / 3 - 1, dw = tap % 3 - 1;
    const bool ok =
        ((unsigned)(hA + dh) < HH) && ((unsigned)(wA + dw) < WW);
    const char* ga = ok ? (const char*)(aSrc + (dh * WW + dw) * C_CH + ks2 * KC)
                        : (const char*)zpad;
    const char* gb = (const char*)(bSrc + tap * C_CH + ks2 * KC);
#pragma unroll
    for (int c = 0; c < 4; ++c)
      async_ld16(bufB + aDstB + c * 16, ga + c * 16);
#pragma unroll
    for (int c = 0; c < 4; ++c)
      async_ld16(bufB + bDstB + c * 16, gb + c * 16);
  };

  // Prologue: two stages in flight.
  stage_step(0, 0u);
  stage_step(1, (unsigned)(BUF_E * 2));

  unsigned bufC = 0;  // compute-buffer index (0..2)
  for (int s = 0; s < NSTAGE; ++s) {
    if (s + 1 < NSTAGE) wait_async_le8();  // oldest stage has landed
    else                wait_async0();
    __syncthreads();    // all waves' stage-s data visible; old reads retired
    if (s + 2 < NSTAGE) {
      unsigned bufS = bufC + 2; if (bufS >= 3) bufS -= 3;
      stage_step(s + 2, bufS * (unsigned)(BUF_E * 2));
    }

    const unsigned baseE = bufC * (unsigned)BUF_E;
#pragma unroll
    for (int s2 = 0; s2 < 2; ++s2) {     // two 32-wide WMMA slices per stage
      const unsigned ko = (unsigned)(s2 * 32);
      v16bf afrag[2];
#pragma unroll
      for (int i = 0; i < 2; ++i) {
        const __bf16* p = &sm[baseE + aOffE[i] + ko];
        afrag[i] = cat8(*(const v8bf*)p, *(const v8bf*)(p + 16));
      }
      v16bf bfrag[4];
#pragma unroll
      for (int j = 0; j < 4; ++j) {
        const __bf16* p = &sm[baseE + bOffE[j] + ko];
        bfrag[j] = cat8(*(const v8bf*)p, *(const v8bf*)(p + 16));
      }
#pragma unroll
      for (int i = 0; i < 2; ++i)
#pragma unroll
        for (int j = 0; j < 4; ++j)
          acc[i][j] = __builtin_amdgcn_wmma_f32_16x16x32_bf16(
              false, afrag[i], false, bfrag[j], (short)0, acc[i][j], false,
              false);
    }
    bufC = (bufC == 2) ? 0u : bufC + 1u;
  }

  // Epilogue. C/D layout: VGPR v holds M = v (lanes 0-15) or v+8 (lanes 16-31),
  // N = lane & 15.
  const int rowAdd = (lane >> 4) << 3;
#pragma unroll
  for (int i = 0; i < 2; ++i) {
#pragma unroll
    for (int j = 0; j < 4; ++j) {
      const int cidx = colBase + j * 16 + l16;
      const float bv = bias[cidx];
#pragma unroll
      for (int v = 0; v < 8; ++v) {
        const int m = rowBase + i * 16 + rowAdd + v;
        float val = acc[i][j][v] + bv;
        if (MODE == 1) {
          val = fmaxf(val, 0.0f);
          outbf[(size_t)m * C_CH + cidx] = (__bf16)val;
        } else {
          const int nimg = m / HWP;
          const int hw   = m - nimg * HWP;
          const size_t o = ((size_t)nimg * C_CH + cidx) * HWP + hw;
          val += resid[o];
          val = fmaxf(val, 0.0f);
          outf[o] = val;
        }
      }
    }
  }
}

extern "C" void kernel_launch(void* const* d_in, const int* in_sizes, int n_in,
                              void* d_out, int out_size, void* d_ws,
                              size_t ws_size, hipStream_t stream) {
  const float* x   = (const float*)d_in[0];
  const float* w1  = (const float*)d_in[1];
  const float* g1  = (const float*)d_in[2];
  const float* b1  = (const float*)d_in[3];
  const float* rm1 = (const float*)d_in[4];
  const float* rv1 = (const float*)d_in[5];
  const float* w2  = (const float*)d_in[6];
  const float* g2  = (const float*)d_in[7];
  const float* b2  = (const float*)d_in[8];
  const float* rm2 = (const float*)d_in[9];
  const float* rv2 = (const float*)d_in[10];

  char* ws = (char*)d_ws;
  const size_t sizeXb = (size_t)MTOT * C_CH * 2;   // 51,380,224 B
  const size_t sizeWb = (size_t)C_CH * KK * 2;     //  1,179,648 B
  __bf16* xb    = (__bf16*)(ws);
  __bf16* yb    = (__bf16*)(ws + sizeXb);
  __bf16* wb1   = (__bf16*)(ws + 2 * sizeXb);
  __bf16* wb2   = (__bf16*)(ws + 2 * sizeXb + sizeWb);
  float*  bias1 = (float*)(ws + 2 * sizeXb + 2 * sizeWb);
  float*  bias2 = (float*)(ws + 2 * sizeXb + 2 * sizeWb + C_CH * sizeof(float));
  float*  zpad  = (float*)(ws + 2 * sizeXb + 2 * sizeWb + 2 * C_CH * sizeof(float));

  const int wBlocks = (C_CH * KK + 255) / 256;     // 2304
  prep_w<<<wBlocks, 256, 0, stream>>>(w1, g1, b1, rm1, rv1, wb1, bias1);
  prep_w<<<wBlocks, 256, 0, stream>>>(w2, g2, b2, rm2, rv2, wb2, bias2);

  cvt_x<<<MTOT, 256, 0, stream>>>(x, xb, zpad);    // MTOT*C/256 = MTOT blocks

  dim3 grid(MTOT / 128, C_CH / 128);               // 784 x 2
  conv_gemm<1><<<grid, 256, SMEM_BYTES, stream>>>(xb, wb1, bias1, nullptr,
                                                  zpad, yb, nullptr);
  conv_gemm<2><<<grid, 256, SMEM_BYTES, stream>>>(yb, wb2, bias2, x, zpad,
                                                  nullptr, (float*)d_out);
}
